// KmeansCosineQuantizer_6760278524432
// MI455X (gfx1250) — matrix operand order repaired and checked by reading, not compiled, not run on hardware
//
#include <hip/hip_runtime.h>
#include <cstdint>

// ---- geometry ----
#define D_        512
#define CTILE     16           // centroids per staged tile
#define ROWS_WAVE 32           // 2 resident A panels of 16 rows per wave
#define WAVES     8
#define ROWS_WG   (ROWS_WAVE * WAVES)   // 256 rows per workgroup
#define THREADS   256
#define XSTRIDE   520          // bf16 elems per LDS row (512 + 8 pad)
#define TILE_BYTES (CTILE * D_ * 2)     // 16 KB of bf16 per cb tile (unpadded, in ws)

#if __has_builtin(__builtin_amdgcn_tensor_load_to_lds) && __has_builtin(__builtin_amdgcn_s_wait_tensorcnt)
#define HAVE_TDM 1
#else
#define HAVE_TDM 0
#endif

typedef __attribute__((ext_vector_type(16))) __bf16        v16bf;
typedef __attribute__((ext_vector_type(8)))  float         v8f;
typedef __attribute__((ext_vector_type(8)))  unsigned int  v8u;
typedef __attribute__((ext_vector_type(4)))  unsigned int  u32x4;
typedef __attribute__((ext_vector_type(4)))  float         f32x4;
typedef __attribute__((ext_vector_type(8)))  int           i32x8;
typedef __attribute__((ext_vector_type(4)))  int           i32x4;

// round-to-nearest-even f32 -> bf16, pack two into one dword (lo in [15:0])
static __device__ __forceinline__ unsigned pack_bf16(float lo, float hi) {
    unsigned a = __builtin_bit_cast(unsigned, lo);
    unsigned b = __builtin_bit_cast(unsigned, hi);
    a = (a + 0x7FFFu + ((a >> 16) & 1u)) >> 16;
    b = (b + 0x7FFFu + ((b >> 16) & 1u)) & 0xFFFF0000u;
    return a | b;
}

// ---- pre-pass: codebook f32 -> packed bf16 in workspace ----
__global__ void cvt_cb_bf16(const float* __restrict__ cb,
                            unsigned* __restrict__ out, int n2) {
    int i = blockIdx.x * blockDim.x + threadIdx.x;    // one packed dword (2 elems)
    if (i < n2) {
        float lo = cb[2 * i];
        float hi = cb[2 * i + 1];
        out[i] = pack_bf16(lo, hi);
    }
}

#if HAVE_TDM
// Issue one TDM copy: 16 rows x 512 bf16 tile (contiguous in ws) -> LDS,
// with hardware padding of 4 DWORDs every 256 DWORDs => LDS row stride = 520 bf16.
// D# fields per CDNA5 ISA.
// This toolchain exposes the 6-arg builtin:
//   (uint32x4 g0, int32x8 g1, int32x4 g2, int32x4 g3, int32x8 g4, i32 cpol)
// Our tile is 2D, so descriptor groups beyond 0/1 are all-zero (count=1 in g0).
static __device__ __forceinline__ void tdm_load_tile(const unsigned short* srcTile,
                                                     unsigned ldsByteOff) {
    unsigned long long ga = (unsigned long long)(uintptr_t)srcTile;
    u32x4 g0;
    g0[0] = 1u;                                     // count=1, user descriptor
    g0[1] = ldsByteOff;                             // lds_addr (bytes)
    g0[2] = (unsigned)(ga & 0xFFFFFFFFu);           // global_addr[31:0]
    g0[3] = (unsigned)((ga >> 32) & 0x01FFFFFFu)    // global_addr[56:32]
          | (2u << 30);                             // type = 2 ("image")
    i32x8 g1;
    g1[0] = (int)((1u << 16)      // data_size = 2 bytes
                | (1u << 20)      // pad_enable
                | (7u << 22)      // pad_interval: every 256 DWORDs (1024 B = one row)
                | (3u << 25));    // pad_amount: 4 DWORDs (16 B)
    g1[1] = (int)(512u << 16);    // tensor_dim0 = 512 (atomic_barrier_addr = 0)
    g1[2] = (int)(16u << 16);     // tensor_dim0 hi | tensor_dim1 = 16
    g1[3] = (int)(512u << 16);    // tensor_dim1 hi | tile_dim0 = 512
    g1[4] = 16;                   // tile_dim1 = 16, tile_dim2 = 0
    g1[5] = 512;                  // tensor_dim0_stride = 512 elems
    g1[6] = 0;                    // stride hi | tensor_dim1_stride lo (unused, 2D)
    g1[7] = 0;
    i32x4 z4 = {0, 0, 0, 0};
    i32x8 z8 = {0, 0, 0, 0, 0, 0, 0, 0};
    __builtin_amdgcn_tensor_load_to_lds(g0, g1, z4, z4, z8, 0);
}
#endif

// =====================================================================
// Main fused kernel: TDM-staged bf16 codebook from workspace.
// =====================================================================
__global__ __launch_bounds__(THREADS, 1)
void kmeans_fused_tdm(const float* __restrict__ x,
                      const unsigned short* __restrict__ cbh,  // [K,512] bf16 (ws)
                      const float* __restrict__ cb,            // [K,512] f32 (exact preds)
                      float* __restrict__ preds,
                      long long* __restrict__ labels,
                      float* __restrict__ sims,
                      int K)
{
    __shared__ unsigned short bt[2][CTILE * XSTRIDE];
    __shared__ int lab_s[ROWS_WG];

    const int tid  = threadIdx.x;
    const int wave = tid >> 5;
    const int lane = tid & 31;
    const int l16  = lane & 15;
    const int hi   = lane >> 4;
    const int rowBase = blockIdx.x * ROWS_WG;

    // ---- resident A panels: 2 x (16 rows x 512) bf16 fragments in VGPRs ----
    v16bf afrag0[D_ / 32];
    v16bf afrag1[D_ / 32];
    {
        const int kb = hi * 8;
#pragma unroll
        for (int g = 0; g < 2; ++g) {
            const float* xrow =
                x + (size_t)(rowBase + wave * ROWS_WAVE + g * 16 + l16) * D_;
#pragma unroll
            for (int s = 0; s < D_ / 32; ++s) {
                f32x4 a0 = *(const f32x4*)(xrow + s * 32 + kb);
                f32x4 a1 = *(const f32x4*)(xrow + s * 32 + kb + 4);
                f32x4 a2 = *(const f32x4*)(xrow + s * 32 + kb + 16);
                f32x4 a3 = *(const f32x4*)(xrow + s * 32 + kb + 20);
                v8u u;
                u[0] = pack_bf16(a0[0], a0[1]); u[1] = pack_bf16(a0[2], a0[3]);
                u[2] = pack_bf16(a1[0], a1[1]); u[3] = pack_bf16(a1[2], a1[3]);
                u[4] = pack_bf16(a2[0], a2[1]); u[5] = pack_bf16(a2[2], a2[3]);
                u[6] = pack_bf16(a3[0], a3[1]); u[7] = pack_bf16(a3[2], a3[3]);
                if (g == 0) afrag0[s] = __builtin_bit_cast(v16bf, u);
                else        afrag1[s] = __builtin_bit_cast(v16bf, u);
            }
        }
    }

    float bv0[8], bv1[8];
    int   bi0[8], bi1[8];
#pragma unroll
    for (int r = 0; r < 8; ++r) {
        bv0[r] = bv1[r] = -3.402823466e38f;
        bi0[r] = bi1[r] = 0;
    }

    const int NT = K / CTILE;

    // ---- prologue: stage tile 0 ----
#if HAVE_TDM
    if (wave == 0)
        tdm_load_tile(cbh, (unsigned)(uintptr_t)&bt[0][0]);
    __builtin_amdgcn_s_wait_tensorcnt(0);
#else
    for (int it = 0; it < (TILE_BYTES / 16) / THREADS; ++it) {   // 4 x u32x4 per thread
        int idx = it * THREADS + tid;                            // 16B chunk index
        int r   = idx >> 6;                                      // 64 chunks per row
        int c16 = idx & 63;
        u32x4 v = *(const u32x4*)(cbh + (size_t)r * D_ + c16 * 8);
        *(u32x4*)&bt[0][r * XSTRIDE + c16 * 8] = v;
    }
#endif
    __syncthreads();

    // ---- main loop: async staging of t+1 overlaps WMMA on t ----
    for (int t = 0; t < NT; ++t) {
#if HAVE_TDM
        if (wave == 0 && t + 1 < NT)
            tdm_load_tile(cbh + (size_t)(t + 1) * CTILE * D_,
                          (unsigned)(uintptr_t)&bt[(t + 1) & 1][0]);
#else
        const int tn = (t + 1 < NT) ? t + 1 : 0;
        u32x4 pf[4];
#pragma unroll
        for (int it = 0; it < 4; ++it) {
            int idx = it * THREADS + tid;
            int r   = idx >> 6;
            int c16 = idx & 63;
            pf[it] = *(const u32x4*)(cbh + (size_t)(tn * CTILE + r) * D_ + c16 * 8);
        }
#endif

        // B layout (16-bit B 32x16): lanes 0-15: K 0..7,16..23, lanes 16-31: K 8..15,24..31
        const unsigned short* brow = &bt[t & 1][l16 * XSTRIDE];
        v8f c0 = {0.f, 0.f, 0.f, 0.f, 0.f, 0.f, 0.f, 0.f};
        v8f c1 = {0.f, 0.f, 0.f, 0.f, 0.f, 0.f, 0.f, 0.f};
#pragma unroll
        for (int s = 0; s < D_ / 32; ++s) {
            u32x4 b0 = *(const u32x4*)(brow + s * 32 + hi * 8);
            u32x4 b1 = *(const u32x4*)(brow + s * 32 + hi * 8 + 16);
            v8u u;
            u[0] = b0[0]; u[1] = b0[1]; u[2] = b0[2]; u[3] = b0[3];
            u[4] = b1[0]; u[5] = b1[1]; u[6] = b1[2]; u[7] = b1[3];
            v16bf bfrag = __builtin_bit_cast(v16bf, u);
            c0 = __builtin_amdgcn_wmma_f32_16x16x32_bf16(
                     false, afrag0[s], false, bfrag, (short)0, c0, false, false);
            c1 = __builtin_amdgcn_wmma_f32_16x16x32_bf16(
                     false, afrag1[s], false, bfrag, (short)0, c1, false, false);
        }

        const int col = t * CTILE + l16;
#pragma unroll
        for (int r = 0; r < 8; ++r) {
            int rowg0 = rowBase + wave * ROWS_WAVE + hi * 8 + r;
            int rowg1 = rowg0 + 16;
            sims[(size_t)rowg0 * K + col] = c0[r];
            sims[(size_t)rowg1 * K + col] = c1[r];
            bool g0 = c0[r] > bv0[r];
            bv0[r] = g0 ? c0[r] : bv0[r];
            bi0[r] = g0 ? col   : bi0[r];
            bool g1 = c1[r] > bv1[r];
            bv1[r] = g1 ? c1[r] : bv1[r];
            bi1[r] = g1 ? col   : bi1[r];
        }

#if HAVE_TDM
        __builtin_amdgcn_s_wait_tensorcnt(0);
#else
#pragma unroll
        for (int it = 0; it < 4; ++it) {
            int idx = it * THREADS + tid;
            int r   = idx >> 6;
            int c16 = idx & 63;
            *(u32x4*)&bt[(t + 1) & 1][r * XSTRIDE + c16 * 8] = pf[it];
        }
#endif
        __syncthreads();
    }

    // ---- argmax reduction across the 16 lane-columns within each half ----
#pragma unroll
    for (int r = 0; r < 8; ++r) {
#pragma unroll
        for (int off = 8; off >= 1; off >>= 1) {
            float ov0 = __shfl_xor(bv0[r], off, 32);
            int   oi0 = __shfl_xor(bi0[r], off, 32);
            if (ov0 > bv0[r] || (ov0 == bv0[r] && oi0 < bi0[r])) {
                bv0[r] = ov0; bi0[r] = oi0;
            }
            float ov1 = __shfl_xor(bv1[r], off, 32);
            int   oi1 = __shfl_xor(bi1[r], off, 32);
            if (ov1 > bv1[r] || (ov1 == bv1[r] && oi1 < bi1[r])) {
                bv1[r] = ov1; bi1[r] = oi1;
            }
        }
        if (l16 == 0) {
            lab_s[wave * ROWS_WAVE + hi * 8 + r]      = bi0[r];
            lab_s[wave * ROWS_WAVE + 16 + hi * 8 + r] = bi1[r];
        }
    }
    __syncthreads();

    // ---- labels (int64) + exact preds gather from the f32 codebook ----
    labels[rowBase + tid] = (long long)lab_s[tid];

#pragma unroll 4
    for (int it = 0; it < (ROWS_WG * (D_ / 4)) / THREADS; ++it) {
        int idx = it * THREADS + tid;
        int r   = idx >> 7;
        int k4  = idx & 127;
        int lbl = lab_s[r];
        f32x4 v = *(const f32x4*)(cb + (size_t)lbl * D_ + k4 * 4);
        *(f32x4*)(preds + (size_t)(rowBase + r) * D_ + k4 * 4) = v;
    }
}

// =====================================================================
// Fallback (no workspace needed): self-contained f32 staging.
// =====================================================================
__global__ __launch_bounds__(THREADS, 1)
void kmeans_fused_fallback(const float* __restrict__ x,
                           const float* __restrict__ cb,
                           float* __restrict__ preds,
                           long long* __restrict__ labels,
                           float* __restrict__ sims,
                           int K)
{
    __shared__ unsigned short bt[2][CTILE * XSTRIDE];
    __shared__ int lab_s[ROWS_WG];

    const int tid  = threadIdx.x;
    const int wave = tid >> 5;
    const int lane = tid & 31;
    const int l16  = lane & 15;
    const int hi   = lane >> 4;
    const int rowBase = blockIdx.x * ROWS_WG;

    v16bf afrag0[D_ / 32];
    v16bf afrag1[D_ / 32];
    {
        const int kb = hi * 8;
#pragma unroll
        for (int g = 0; g < 2; ++g) {
            const float* xrow =
                x + (size_t)(rowBase + wave * ROWS_WAVE + g * 16 + l16) * D_;
#pragma unroll
            for (int s = 0; s < D_ / 32; ++s) {
                f32x4 a0 = *(const f32x4*)(xrow + s * 32 + kb);
                f32x4 a1 = *(const f32x4*)(xrow + s * 32 + kb + 4);
                f32x4 a2 = *(const f32x4*)(xrow + s * 32 + kb + 16);
                f32x4 a3 = *(const f32x4*)(xrow + s * 32 + kb + 20);
                v8u u;
                u[0] = pack_bf16(a0[0], a0[1]); u[1] = pack_bf16(a0[2], a0[3]);
                u[2] = pack_bf16(a1[0], a1[1]); u[3] = pack_bf16(a1[2], a1[3]);
                u[4] = pack_bf16(a2[0], a2[1]); u[5] = pack_bf16(a2[2], a2[3]);
                u[6] = pack_bf16(a3[0], a3[1]); u[7] = pack_bf16(a3[2], a3[3]);
                if (g == 0) afrag0[s] = __builtin_bit_cast(v16bf, u);
                else        afrag1[s] = __builtin_bit_cast(v16bf, u);
            }
        }
    }

    float bv0[8], bv1[8];
    int   bi0[8], bi1[8];
#pragma unroll
    for (int r = 0; r < 8; ++r) {
        bv0[r] = bv1[r] = -3.402823466e38f;
        bi0[r] = bi1[r] = 0;
    }

#pragma unroll
    for (int it = 0; it < (CTILE * (D_ / 4)) / THREADS; ++it) {
        int idx = it * THREADS + tid;
        int r   = idx >> 7;
        int k4  = idx & 127;
        f32x4 v = *(const f32x4*)(cb + (size_t)r * D_ + k4 * 4);
        unsigned* dst = (unsigned*)&bt[0][r * XSTRIDE + k4 * 4];
        dst[0] = pack_bf16(v[0], v[1]);
        dst[1] = pack_bf16(v[2], v[3]);
    }
    __syncthreads();

    const int NT = K / CTILE;
    for (int t = 0; t < NT; ++t) {
        const int tn = (t + 1 < NT) ? t + 1 : 0;
        f32x4 pf[8];
#pragma unroll
        for (int it = 0; it < 8; ++it) {
            int idx = it * THREADS + tid;
            int r   = idx >> 7;
            int k4  = idx & 127;
            pf[it] = *(const f32x4*)(cb + (size_t)(tn * CTILE + r) * D_ + k4 * 4);
        }

        const unsigned short* brow = &bt[t & 1][l16 * XSTRIDE];
        v8f c0 = {0.f, 0.f, 0.f, 0.f, 0.f, 0.f, 0.f, 0.f};
        v8f c1 = {0.f, 0.f, 0.f, 0.f, 0.f, 0.f, 0.f, 0.f};
#pragma unroll
        for (int s = 0; s < D_ / 32; ++s) {
            u32x4 b0 = *(const u32x4*)(brow + s * 32 + hi * 8);
            u32x4 b1 = *(const u32x4*)(brow + s * 32 + hi * 8 + 16);
            v8u u;
            u[0] = b0[0]; u[1] = b0[1]; u[2] = b0[2]; u[3] = b0[3];
            u[4] = b1[0]; u[5] = b1[1]; u[6] = b1[2]; u[7] = b1[3];
            v16bf bfrag = __builtin_bit_cast(v16bf, u);
            c0 = __builtin_amdgcn_wmma_f32_16x16x32_bf16(
                     false, afrag0[s], false, bfrag, (short)0, c0, false, false);
            c1 = __builtin_amdgcn_wmma_f32_16x16x32_bf16(
                     false, afrag1[s], false, bfrag, (short)0, c1, false, false);
        }

        const int col = t * CTILE + l16;
#pragma unroll
        for (int r = 0; r < 8; ++r) {
            int rowg0 = rowBase + wave * ROWS_WAVE + hi * 8 + r;
            int rowg1 = rowg0 + 16;
            sims[(size_t)rowg0 * K + col] = c0[r];
            sims[(size_t)rowg1 * K + col] = c1[r];
            bool g0 = c0[r] > bv0[r];
            bv0[r] = g0 ? c0[r] : bv0[r];
            bi0[r] = g0 ? col   : bi0[r];
            bool g1 = c1[r] > bv1[r];
            bv1[r] = g1 ? c1[r] : bv1[r];
            bi1[r] = g1 ? col   : bi1[r];
        }

#pragma unroll
        for (int it = 0; it < 8; ++it) {
            int idx = it * THREADS + tid;
            int r   = idx >> 7;
            int k4  = idx & 127;
            unsigned* dst = (unsigned*)&bt[(t + 1) & 1][r * XSTRIDE + k4 * 4];
            dst[0] = pack_bf16(pf[it][0], pf[it][1]);
            dst[1] = pack_bf16(pf[it][2], pf[it][3]);
        }
        __syncthreads();
    }

#pragma unroll
    for (int r = 0; r < 8; ++r) {
#pragma unroll
        for (int off = 8; off >= 1; off >>= 1) {
            float ov0 = __shfl_xor(bv0[r], off, 32);
            int   oi0 = __shfl_xor(bi0[r], off, 32);
            if (ov0 > bv0[r] || (ov0 == bv0[r] && oi0 < bi0[r])) {
                bv0[r] = ov0; bi0[r] = oi0;
            }
            float ov1 = __shfl_xor(bv1[r], off, 32);
            int   oi1 = __shfl_xor(bi1[r], off, 32);
            if (ov1 > bv1[r] || (ov1 == bv1[r] && oi1 < bi1[r])) {
                bv1[r] = ov1; bi1[r] = oi1;
            }
        }
        if (l16 == 0) {
            lab_s[wave * ROWS_WAVE + hi * 8 + r]      = bi0[r];
            lab_s[wave * ROWS_WAVE + 16 + hi * 8 + r] = bi1[r];
        }
    }
    __syncthreads();

    labels[rowBase + tid] = (long long)lab_s[tid];

#pragma unroll 4
    for (int it = 0; it < (ROWS_WG * (D_ / 4)) / THREADS; ++it) {
        int idx = it * THREADS + tid;
        int r   = idx >> 7;
        int k4  = idx & 127;
        int lbl = lab_s[r];
        f32x4 v = *(const f32x4*)(cb + (size_t)lbl * D_ + k4 * 4);
        *(f32x4*)(preds + (size_t)(rowBase + r) * D_ + k4 * 4) = v;
    }
}

extern "C" void kernel_launch(void* const* d_in, const int* in_sizes, int n_in,
                              void* d_out, int out_size, void* d_ws, size_t ws_size,
                              hipStream_t stream) {
    (void)n_in; (void)out_size;
    const float* x  = (const float*)d_in[0];          // [N, 512] f32
    const float* cb = (const float*)d_in[1];          // [K, 512] f32
    const int N = in_sizes[0] / D_;
    const int K = in_sizes[1] / D_;

    // d_out flat layout (return order): preds [N,512] f32, labels [N] i64, sims [N,K] f32
    char* base = (char*)d_out;
    float*     preds  = (float*)base;
    long long* labels = (long long*)(base + (size_t)N * D_ * sizeof(float));
    float*     sims   = (float*)(base + (size_t)N * D_ * sizeof(float)
                                      + (size_t)N * sizeof(long long));

    dim3 grid(N / ROWS_WG);
    dim3 block(THREADS);

    const size_t need = (size_t)K * D_ * sizeof(unsigned short);  // bf16 codebook
    if (d_ws != nullptr && ws_size >= need) {
        // 1) convert codebook to bf16 in workspace, 2) TDM-staged fused kernel
        unsigned* cbh = (unsigned*)d_ws;
        int n2 = K * D_ / 2;
        cvt_cb_bf16<<<dim3((n2 + THREADS - 1) / THREADS), block, 0, stream>>>(cb, cbh, n2);
        kmeans_fused_tdm<<<grid, block, 0, stream>>>(
            x, (const unsigned short*)cbh, cb, preds, labels, sims, K);
    } else {
        kmeans_fused_fallback<<<grid, block, 0, stream>>>(x, cb, preds, labels, sims, K);
    }
}
